// SingleHeadGATLayer_87866440941694
// MI455X (gfx1250) — matrix operand order repaired and checked by reading, not compiled, not run on hardware
//
#include <hip/hip_runtime.h>
#include <hip/hip_bf16.h>
#include <math.h>

#define N_NODES 100000
#define N_EDGES 1600000
#define DIM     128
#define SLOPE   0.01f

typedef __attribute__((ext_vector_type(16))) _Float16 v16h;
typedef __attribute__((ext_vector_type(8)))  float    v8f;
typedef __attribute__((ext_vector_type(4)))  float    v4f;

// ---------------------------------------------------------------------------
// Kernel 1: z = X @ W^T via v_wmma_f32_16x16x32_f16.
// Each wave computes a 16-row x 128-col strip of Z (8 accumulators).
// W (128x128 f32 = 64KB) is staged into LDS as f16 once per block.
// ---------------------------------------------------------------------------
__global__ void __launch_bounds__(256)
gat_gemm_wmma(const float* __restrict__ X, const float* __restrict__ W,
              float* __restrict__ Z, int nStrips)
{
    __shared__ __align__(32) _Float16 Wlds[DIM * DIM];  // 32 KB, row-major [o][k]
    for (int i = threadIdx.x; i < DIM * DIM; i += blockDim.x)
        Wlds[i] = (_Float16)W[i];
    __syncthreads();

    const int wavesPerBlk = blockDim.x >> 5;
    const int strip = blockIdx.x * wavesPerBlk + (threadIdx.x >> 5);
    if (strip >= nStrips) return;           // wave-uniform: EXEC stays all-1s
    const int lane = threadIdx.x & 31;
    const int r    = lane & 15;             // row (A) / col (B,C)
    const int h    = lane >> 4;             // K-half selector
    const int n0   = strip * 16;

    const float* arow = X + (size_t)(n0 + r) * DIM;

    const v8f vzero = {0.f, 0.f, 0.f, 0.f, 0.f, 0.f, 0.f, 0.f};
    v8f acc[8];
#pragma unroll
    for (int ct = 0; ct < 8; ++ct) acc[ct] = vzero;

#pragma unroll
    for (int kb = 0; kb < 4; ++kb) {
        const int k0 = kb * 32;
        // A fragment (16x32 f16): lane l -> row l&15;
        //   elems 0..15 : K = k0 + h*16 + e
        v4f f0 = *(const v4f*)(arow + k0 + h * 16);
        v4f f1 = *(const v4f*)(arow + k0 + h * 16 + 4);
        v4f f2 = *(const v4f*)(arow + k0 + h * 16 + 8);
        v4f f3 = *(const v4f*)(arow + k0 + h * 16 + 12);
        v16h a;
#pragma unroll
        for (int e = 0; e < 4; ++e) {
            a[e]      = (_Float16)f0[e];
            a[e + 4]  = (_Float16)f1[e];
            a[e + 8]  = (_Float16)f2[e];
            a[e + 12] = (_Float16)f3[e];
        }
#pragma unroll
        for (int ct = 0; ct < 8; ++ct) {
            // B fragment (32x16 f16): lane l -> col o0 + (l&15),
            // 16 contiguous K values starting at k0 + h*16 (one 32B LDS load).
            const v16h b = *(const v16h*)(Wlds + (size_t)(ct * 16 + r) * DIM + k0 + h * 16);
            acc[ct] = __builtin_amdgcn_wmma_f32_16x16x32_f16(
                false, a, false, b, (short)0, acc[ct], false, false);
        }
    }

    // C/D layout: element (v, lane) -> row v + 8*h, col r
#pragma unroll
    for (int ct = 0; ct < 8; ++ct) {
        float* zp = Z + (size_t)(n0 + h * 8) * DIM + ct * 16 + r;
#pragma unroll
        for (int v = 0; v < 8; ++v)
            zp[(size_t)v * DIM] = acc[ct][v];
    }
}

// ---------------------------------------------------------------------------
// Kernel 2: init out=0, m=-inf, denom=0
// ---------------------------------------------------------------------------
__global__ void __launch_bounds__(256)
gat_init(float* __restrict__ out, float* __restrict__ m, float* __restrict__ denom,
         int n, int total)
{
    int i = blockIdx.x * blockDim.x + threadIdx.x;
    if (i < total) out[i] = 0.f;
    if (i < n) { m[i] = -INFINITY; denom[i] = 0.f; }
}

// ---------------------------------------------------------------------------
// Kernel 3: es[n] = z[n] . a_src ; ed[n] = z[n] . a_dst  (one wave per node)
// ---------------------------------------------------------------------------
__global__ void __launch_bounds__(256)
gat_node_scores(const float* __restrict__ Z, const float* __restrict__ a_attn,
                float* __restrict__ es, float* __restrict__ ed, int n)
{
    int wid = blockIdx.x * (blockDim.x >> 5) + (threadIdx.x >> 5);
    if (wid >= n) return;
    int lane = threadIdx.x & 31;
    v4f zv = *(const v4f*)(Z + (size_t)wid * DIM + lane * 4);
    v4f as = *(const v4f*)(a_attn + lane * 4);
    v4f ad = *(const v4f*)(a_attn + DIM + lane * 4);
    float s = zv[0]*as[0] + zv[1]*as[1] + zv[2]*as[2] + zv[3]*as[3];
    float d = zv[0]*ad[0] + zv[1]*ad[1] + zv[2]*ad[2] + zv[3]*ad[3];
#pragma unroll
    for (int off = 16; off; off >>= 1) {
        s += __shfl_xor(s, off, 32);
        d += __shfl_xor(d, off, 32);
    }
    if (lane == 0) { es[wid] = s; ed[wid] = d; }
}

// float atomic max via monotonic int mapping (m initialized to -inf)
__device__ inline void atomicMaxF(float* addr, float v)
{
    if (v >= 0.f)
        __hip_atomic_fetch_max((int*)addr, __float_as_int(v),
                               __ATOMIC_RELAXED, __HIP_MEMORY_SCOPE_AGENT);
    else
        __hip_atomic_fetch_min((unsigned int*)addr, __float_as_uint(v),
                               __ATOMIC_RELAXED, __HIP_MEMORY_SCOPE_AGENT);
}

// ---------------------------------------------------------------------------
// Kernel 4: per-edge score e = leaky_relu(es[src]+ed[dst]); segment max into m
// ---------------------------------------------------------------------------
__global__ void __launch_bounds__(256)
gat_edge_score(const int* __restrict__ src, const int* __restrict__ dst,
               const float* __restrict__ es, const float* __restrict__ ed,
               float* __restrict__ ebuf, float* __restrict__ m, int e)
{
    int i = blockIdx.x * blockDim.x + threadIdx.x;
    if (i >= e) return;
    int s = src[i], d = dst[i];
    float ev = es[s] + ed[d];
    ev = (ev > 0.f) ? ev : SLOPE * ev;
    ebuf[i] = ev;
    atomicMaxF(&m[d], ev);
}

// ---------------------------------------------------------------------------
// Kernel 5: ex = exp(e - m[dst]); denom[dst] += ex  (ebuf overwritten with ex)
// ---------------------------------------------------------------------------
__global__ void __launch_bounds__(256)
gat_edge_expsum(const int* __restrict__ dst, float* __restrict__ ebuf,
                const float* __restrict__ m, float* __restrict__ denom, int e)
{
    int i = blockIdx.x * blockDim.x + threadIdx.x;
    if (i >= e) return;
    int d = dst[i];
    float ex = expf(ebuf[i] - m[d]);
    ebuf[i] = ex;
    __hip_atomic_fetch_add(&denom[d], ex, __ATOMIC_RELAXED, __HIP_MEMORY_SCOPE_AGENT);
}

// ---------------------------------------------------------------------------
// Kernel 6: h[dst] += (ex/denom[dst]) * z[src]   (one wave per edge, float4/lane)
// ---------------------------------------------------------------------------
__global__ void __launch_bounds__(256)
gat_edge_aggregate(const int* __restrict__ src, const int* __restrict__ dst,
                   const float* __restrict__ Z, const float* __restrict__ ebuf,
                   const float* __restrict__ denom, float* __restrict__ out, int e)
{
    int wid = blockIdx.x * (blockDim.x >> 5) + (threadIdx.x >> 5);
    if (wid >= e) return;
    int lane = threadIdx.x & 31;
    int s = src[wid], d = dst[wid];
    float w = ebuf[wid] / denom[d];
    v4f zv = *(const v4f*)(Z + (size_t)s * DIM + lane * 4);
    float* op = out + (size_t)d * DIM + lane * 4;
#pragma unroll
    for (int c = 0; c < 4; ++c)
        __hip_atomic_fetch_add(op + c, w * zv[c],
                               __ATOMIC_RELAXED, __HIP_MEMORY_SCOPE_AGENT);
}

// ---------------------------------------------------------------------------
// Kernel 7: out = elu(out) in place
// ---------------------------------------------------------------------------
__global__ void __launch_bounds__(256)
gat_elu(float* __restrict__ out, int total)
{
    int i = blockIdx.x * blockDim.x + threadIdx.x;
    if (i >= total) return;
    float x = out[i];
    out[i] = (x > 0.f) ? x : (expf(x) - 1.f);
}

// ---------------------------------------------------------------------------
extern "C" void kernel_launch(void* const* d_in, const int* in_sizes, int n_in,
                              void* d_out, int out_size, void* d_ws, size_t ws_size,
                              hipStream_t stream)
{
    const float* features = (const float*)d_in[0];   // N x 128
    const int*   src      = (const int*)d_in[1];     // E
    const int*   dst      = (const int*)d_in[2];     // E
    const float* W_fc     = (const float*)d_in[3];   // 128 x 128
    const float* a_attn   = (const float*)d_in[4];   // 256
    float*       out      = (float*)d_out;           // N x 128

    const int N = in_sizes[0] / DIM;
    const int E = in_sizes[1];

    // workspace layout (floats)
    float* Z     = (float*)d_ws;                     // N*128
    float* es    = Z + (size_t)N * DIM;              // N
    float* ed    = es + N;                           // N
    float* m     = ed + N;                           // N
    float* denom = m + N;                            // N
    float* ebuf  = denom + N;                        // E

    const int nStrips = N / 16;                      // 100000/16 = 6250 exact
    const int total   = N * DIM;

    // 1) GEMM (WMMA): 8 waves/block, one 16x128 strip per wave
    gat_gemm_wmma<<<(nStrips + 7) / 8, 256, 0, stream>>>(features, W_fc, Z, nStrips);

    // 2) init out/m/denom
    gat_init<<<(total + 255) / 256, 256, 0, stream>>>(out, m, denom, N, total);

    // 3) per-node attention scores (one wave per node)
    gat_node_scores<<<(N + 7) / 8, 256, 0, stream>>>(Z, a_attn, es, ed, N);

    // 4) edge scores + segment max
    gat_edge_score<<<(E + 255) / 256, 256, 0, stream>>>(src, dst, es, ed, ebuf, m, E);

    // 5) exp + segment sum
    gat_edge_expsum<<<(E + 255) / 256, 256, 0, stream>>>(dst, ebuf, m, denom, E);

    // 6) weighted aggregation (one wave per edge)
    gat_edge_aggregate<<<(E + 7) / 8, 256, 0, stream>>>(src, dst, Z, ebuf, denom, out, E);

    // 7) ELU
    gat_elu<<<(total + 255) / 256, 256, 0, stream>>>(out, total);
}
